// AtomIoUNet_1451698946674
// MI455X (gfx1250) — matrix operand order repaired and checked by reading, not compiled, not run on hardware
//
#include <hip/hip_runtime.h>

// ---------------------------------------------------------------------------
// CDNA5 (gfx1250) WMMA types
// ---------------------------------------------------------------------------
typedef __attribute__((ext_vector_type(16))) _Float16 v16h;
typedef __attribute__((ext_vector_type(8)))  _Float16 v8h;
typedef __attribute__((ext_vector_type(8)))  float    v8f;

#define LDT 40   // LDS row pitch in f16 (32 used + 8 pad; 80B keeps 16B align, staggers banks)

// Build a v16h fragment from two 16-byte LDS chunks (two ds_load_b128).
__device__ __forceinline__ v16h frag_cat(const _Float16* p0, const _Float16* p1)
{
    v8h lo = *(const v8h*)p0;
    v8h hi = *(const v8h*)p1;
    return __builtin_shufflevector(lo, hi, 0, 1, 2, 3, 4, 5, 6, 7,
                                   8, 9, 10, 11, 12, 13, 14, 15);
}

// ---------------------------------------------------------------------------
// Layout-prep kernels (one-shot, bandwidth-bound; trivial vs 23.3 TB/s HBM)
// ---------------------------------------------------------------------------
// fp32 NCHW -> f16 padded NHWC [Nb][H+2][W+2][C] with zero halo.
__global__ __launch_bounds__(256)
void pad_nhwc_f16(const float* __restrict__ in, _Float16* __restrict__ out,
                  int C, int H, int W, int total)
{
    const int t = blockIdx.x * blockDim.x + threadIdx.x;
    if (t >= total) return;
    const int c  = t % C;
    const int px = t / C;
    const int x  = px % (W + 2);
    const int t2 = px / (W + 2);
    const int y  = t2 % (H + 2);
    const int b  = t2 / (H + 2);
    const int xi = x - 1, yi = y - 1;
    float v = 0.f;
    if (xi >= 0 && xi < W && yi >= 0 && yi < H)
        v = in[((long)(b * C + c) * H + yi) * W + xi];
    out[t] = (_Float16)v;
}

// conv weight OIHW fp32 [Cout][Cin][3][3] -> f16 [3][Cout][3*Cin]
// (k = kx*Cin + ci contiguous per ky: matches padded-NHWC row contiguity)
__global__ __launch_bounds__(256)
void repack_w3_f16(const float* __restrict__ w, _Float16* __restrict__ out,
                   int Cout, int Cin, int total)
{
    const int t = blockIdx.x * blockDim.x + threadIdx.x;
    if (t >= total) return;
    const int ci = t % Cin;
    const int t1 = t / Cin;
    const int kx = t1 % 3;
    const int t2 = t1 / 3;
    const int m  = t2 % Cout;
    const int ky = t2 / Cout;
    out[t] = (_Float16)w[((long)m * Cin + ci) * 9 + ky * 3 + kx];
}

__global__ __launch_bounds__(256)
void cvt_f16(const float* __restrict__ in, _Float16* __restrict__ out, int total)
{
    const int t = blockIdx.x * blockDim.x + threadIdx.x;
    if (t < total) out[t] = (_Float16)in[t];
}

__global__ __launch_bounds__(256)
void zero_f16(_Float16* __restrict__ out, int total)
{
    const int t = blockIdx.x * blockDim.x + threadIdx.x;
    if (t < total) out[t] = (_Float16)0.f;
}

// ---------------------------------------------------------------------------
// 3x3 SAME conv + bias + BN + ReLU as 3 shifted GEMMs (K = 3*Cin contiguous)
// over padded f16 NHWC.  Workgroup (8 waves) computes 64(M) x 128(N) tile.
// Double-buffered LDS; staging = aligned b128 global loads + b128 LDS stores
// with pointer-increment-only addressing; each wave loads 1 A frag + 4 B
// frags (ds_load_b128 pairs, hoisted) and issues 4 WMMAs per 32-wide K step.
// ---------------------------------------------------------------------------
__global__ __launch_bounds__(256)
void conv3x3_bn_relu_wmma(const _Float16* __restrict__ inp,   // [Nb][H+2][W+2][Cin]
                          const _Float16* __restrict__ wrep,  // [3][Cout][3*Cin]
                          const float* __restrict__ bias, const float* __restrict__ scl,
                          const float* __restrict__ off,
                          float* __restrict__ out_nchw,       // nullable
                          _Float16* __restrict__ out_pad,     // nullable [Nb][H+2][W+2][Cout]
                          int Nb, int Cin, int Cout, int H, int W)
{
    __shared__ alignas(16) _Float16 As[2][64 * LDT];
    __shared__ alignas(16) _Float16 Bs[2][128 * LDT];

    const int NP     = Nb * H * W;
    const int tilesN = NP >> 7;
    const int mbase  = (blockIdx.x / tilesN) << 6;
    const int nbase  = (blockIdx.x % tilesN) << 7;

    const int tid  = threadIdx.x;
    const int wave = tid >> 5;
    const int lane = tid & 31;
    const int g    = lane >> 4;
    const int r    = lane & 15;

    // ---- staging roles ----
    const int sRowA = tid >> 2;           // 64 rows x 8 f16  (one v8h)
    const int sKA   = (tid & 3) << 3;
    const int sRowB = tid >> 1;           // 128 rows x 16 f16 (two v8h)
    const int sKB   = (tid & 1) << 4;

    // B staging pixel (fixed): (b, y+ky, x) in padded NHWC, channels contiguous
    const int n_sB = nbase + sRowB;
    const int b_sB = n_sB / (H * W);
    const int rm_B = n_sB - b_sB * (H * W);
    const int y_sB = rm_B / W;
    const int x_sB = rm_B - y_sB * W;

    const int  K3        = 3 * Cin;                     // 384 or 768
    const long aKyStride = (long)Cout * K3;
    const long bKyStride = (long)(W + 2) * Cin;
    const long aRowOff   = (long)(mbase + sRowA) * K3 + sKA;
    const long bRowOff   = (((long)b_sB * (H + 2) + y_sB) * (W + 2) + x_sB) * Cin + sKB;

    // ---- compute roles: wave -> (m subtile, quad of n subtiles) ----
    const int aRow  = ((wave >> 1) << 4) + r;
    const int bRow0 = ((wave & 1) << 6) + r;

    v8f acc[4];
    #pragma unroll
    for (int s = 0; s < 4; ++s) acc[s] = (v8f){0.f,0.f,0.f,0.f,0.f,0.f,0.f,0.f};

    int parity = 0;
    for (int ky = 0; ky < 3; ++ky) {
        const _Float16* Ag = wrep + ky * aKyStride + aRowOff;
        const _Float16* Bg = inp  + ky * bKyStride + bRowOff;

        auto stage = [&](int kb, int buf) {
            *(v8h*)&As[buf][sRowA * LDT + sKA] = *(const v8h*)(Ag + kb);
            _Float16* bs = &Bs[buf][sRowB * LDT + sKB];
            *(v8h*)(bs)     = *(const v8h*)(Bg + kb);
            *(v8h*)(bs + 8) = *(const v8h*)(Bg + kb + 8);
        };

        stage(0, parity);
        for (int kb = 0; kb < K3; kb += 32) {
            __syncthreads();
            if (kb + 32 < K3) stage(kb + 32, parity ^ 1);

            const _Float16* Ab = &As[parity][aRow * LDT];
            const _Float16* Bb = &Bs[parity][bRow0 * LDT];
            // hoist all fragment loads (distinct regs -> overlapped ds waits)
            v16h a  = frag_cat(Ab + g * 16, Ab + g * 16 + 8);
            v16h b0 = frag_cat(Bb +  0 * LDT + g * 16, Bb +  0 * LDT + g * 16 + 8);
            v16h b1 = frag_cat(Bb + 16 * LDT + g * 16, Bb + 16 * LDT + g * 16 + 8);
            v16h b2 = frag_cat(Bb + 32 * LDT + g * 16, Bb + 32 * LDT + g * 16 + 8);
            v16h b3 = frag_cat(Bb + 48 * LDT + g * 16, Bb + 48 * LDT + g * 16 + 8);
            acc[0] = __builtin_amdgcn_wmma_f32_16x16x32_f16(false, a, false, b0,
                                                            (short)0, acc[0], false, false);
            acc[1] = __builtin_amdgcn_wmma_f32_16x16x32_f16(false, a, false, b1,
                                                            (short)0, acc[1], false, false);
            acc[2] = __builtin_amdgcn_wmma_f32_16x16x32_f16(false, a, false, b2,
                                                            (short)0, acc[2], false, false);
            acc[3] = __builtin_amdgcn_wmma_f32_16x16x32_f16(false, a, false, b3,
                                                            (short)0, acc[3], false, false);
            parity ^= 1;
        }
    }

    // ---- epilogue: bias + BN + ReLU; optional NCHW fp32 and padded-NHWC f16 ----
    #pragma unroll
    for (int sub = 0; sub < 4; ++sub) {
        const int n_e  = nbase + ((wave & 1) << 6) + (sub << 4) + r;
        const int b_e  = n_e / (H * W);
        const int rm_e = n_e - b_e * (H * W);
        const int y_e  = rm_e / W;
        const int x_e  = rm_e - y_e * W;
        const long pidx = (((long)b_e * (H + 2) + (y_e + 1)) * (W + 2) + (x_e + 1)) * Cout;
        #pragma unroll
        for (int rr = 0; rr < 8; ++rr) {
            const int m = mbase + ((wave >> 1) << 4) + rr + (g << 3);
            float v = (acc[sub][rr] + bias[m]) * scl[m] + off[m];
            v = fmaxf(v, 0.f);
            if (out_nchw) out_nchw[(((long)b_e * Cout + m) * H + y_e) * W + x_e] = v;
            if (out_pad)  out_pad[pidx + m] = (_Float16)v;
        }
    }
}

// ---------------------------------------------------------------------------
// Dense GEMM + bias + BN + ReLU, all-f16 operands, f16 output:
//   C[n][m] = relu((A[m][:]·B[n][:] + bias)*s+o),  A:[M][K], B:[Nr][K] f16.
// 64x64 tile, double-buffered LDS, 2 WMMAs per wave per 32-wide K step.
// ---------------------------------------------------------------------------
__global__ __launch_bounds__(256)
void gemm_bn_relu_wmma(const _Float16* __restrict__ A, const _Float16* __restrict__ B,
                       const float* __restrict__ bias, const float* __restrict__ scl,
                       const float* __restrict__ off, _Float16* __restrict__ C,
                       int M, int Nr, int K, int ldc)
{
    __shared__ alignas(16) _Float16 As[2][64 * LDT];
    __shared__ alignas(16) _Float16 Bs[2][64 * LDT];

    const int tilesN = Nr >> 6;
    const int mbase  = (blockIdx.x / tilesN) << 6;
    const int nbase  = (blockIdx.x % tilesN) << 6;

    const int tid  = threadIdx.x;
    const int wave = tid >> 5;
    const int lane = tid & 31;
    const int g    = lane >> 4;
    const int r    = lane & 15;

    const int sRow = tid >> 2;
    const int sK   = (tid & 3) << 3;

    const _Float16* Ag = A + (long)(mbase + sRow) * K + sK;
    const _Float16* Bg = B + (long)(nbase + sRow) * K + sK;

    const int aRow  = ((wave >> 1) << 4) + r;
    const int bRow0 = ((wave & 1) << 5) + r;

    v8f acc0 = {0.f, 0.f, 0.f, 0.f, 0.f, 0.f, 0.f, 0.f};
    v8f acc1 = acc0;

    auto stage = [&](int kb, int buf) {
        *(v8h*)&As[buf][sRow * LDT + sK] = *(const v8h*)&Ag[kb];
        *(v8h*)&Bs[buf][sRow * LDT + sK] = *(const v8h*)&Bg[kb];
    };

    stage(0, 0);
    const int nsteps = K >> 5;
    for (int st = 0; st < nsteps; ++st) {
        __syncthreads();
        if (st + 1 < nsteps) {
            stage((st + 1) << 5, (st + 1) & 1);
            __builtin_prefetch(Ag + ((st + 2) << 5), 0, 0);
            __builtin_prefetch(Bg + ((st + 2) << 5), 0, 0);
        }
        const _Float16* Ab  = &As[st & 1][aRow * LDT];
        const _Float16* Bb0 = &Bs[st & 1][bRow0 * LDT];
        const _Float16* Bb1 = Bb0 + 16 * LDT;
        v16h a  = frag_cat(Ab + g * 16, Ab + g * 16 + 8);
        v16h b0 = frag_cat(Bb0 + g * 16, Bb0 + g * 16 + 8);
        v16h b1 = frag_cat(Bb1 + g * 16, Bb1 + g * 16 + 8);
        acc0 = __builtin_amdgcn_wmma_f32_16x16x32_f16(false, a, false, b0,
                                                      (short)0, acc0, false, false);
        acc1 = __builtin_amdgcn_wmma_f32_16x16x32_f16(false, a, false, b1,
                                                      (short)0, acc1, false, false);
    }

    #pragma unroll
    for (int sub = 0; sub < 2; ++sub) {
        const int n_e = nbase + ((wave & 1) << 5) + (sub << 4) + r;
        const v8f& acc = sub ? acc1 : acc0;
        #pragma unroll
        for (int rr = 0; rr < 8; ++rr) {
            const int m = mbase + ((wave >> 1) << 4) + rr + (g << 3);
            float v = (acc[rr] + bias[m]) * scl[m] + off[m];
            v = fmaxf(v, 0.f);
            C[(long)n_e * ldc + m] = (_Float16)v;
        }
    }
}

// ---------------------------------------------------------------------------
// Precise RoI pooling (PrRoIPool2D), fp32 features -> f16 output, optional
// per-(batch,channel) f16 modulation folded in (pooling is linear).
// ---------------------------------------------------------------------------
__device__ __forceinline__ float hat_F(float u)
{
    u = fminf(fmaxf(u, -1.f), 1.f);
    return (u < 0.f) ? 0.5f * (u + 1.f) * (u + 1.f) : 0.5f + u - 0.5f * u * u;
}

__global__ __launch_bounds__(256)
void prroi_pool_kernel(const float* __restrict__ feat, const float* __restrict__ boxes,
                       const _Float16* __restrict__ mod, _Float16* __restrict__ out,
                       int Bf, int C, int H, int W, int P, int ph, int pw,
                       float scl, int row_stride, int col_off, int total)
{
    const int t = blockIdx.x * blockDim.x + threadIdx.x;
    if (t >= total) return;

    const int j = t % pw;
    const int i = (t / pw) % ph;
    const int c = (t / (pw * ph)) % C;
    const int p = (t / (pw * ph * C)) % P;
    const int b = t / (pw * ph * C * P);

    const float* bx = boxes + ((long)b * P + p) * 4;
    const float x1 = bx[0] * scl;
    const float y1 = bx[1] * scl;
    const float x2 = (bx[0] + bx[2]) * scl;
    const float y2 = (bx[1] + bx[3]) * scl;
    const float bw = (x2 - x1) / (float)pw;
    const float bh = (y2 - y1) / (float)ph;

    const float xlo = x1 + j * bw, xhi = xlo + bw;
    const float ylo = y1 + i * bh, yhi = ylo + bh;

    const int h0 = max(0, (int)floorf(ylo) - 1);
    const int h1 = min(H - 1, (int)ceilf(yhi) + 1);
    const int w0 = max(0, (int)floorf(xlo) - 1);
    const int w1 = min(W - 1, (int)ceilf(xhi) + 1);

    const float* frow = feat + ((long)b * C + c) * H * W;
    float acc = 0.f;
    for (int h = h0; h <= h1; ++h) {
        const float wy = hat_F(yhi - (float)h) - hat_F(ylo - (float)h);
        float inner = 0.f;
        for (int wq = w0; wq <= w1; ++wq) {
            const float wx = hat_F(xhi - (float)wq) - hat_F(xlo - (float)wq);
            inner += wx * frow[h * W + wq];
        }
        acc += wy * inner;
    }

    const float area = bw * bh;
    float rv = (area > 0.f) ? acc / fmaxf(area, 1e-8f) : 0.f;
    if (mod) rv *= (float)mod[(long)b * C + c];

    out[((long)b * P + p) * row_stride + col_off + (c * ph + i) * pw + j] = (_Float16)rv;
}

// ---------------------------------------------------------------------------
// IoU head: out[row] = dot(fcbuf[row][0:512] (f16), iou_w) + iou_b
// ---------------------------------------------------------------------------
__global__ __launch_bounds__(256)
void iou_head_kernel(const _Float16* __restrict__ fcbuf, const float* __restrict__ iw,
                     const float* __restrict__ ib, float* __restrict__ out,
                     int rows, int K)
{
    const int t = blockIdx.x * blockDim.x + threadIdx.x;
    if (t >= rows) return;
    const _Float16* rp = fcbuf + (long)t * K;
    float acc = ib[0];
    for (int k = 0; k < K; ++k) acc += (float)rp[k] * iw[k];
    out[t] = acc;
}

// ---------------------------------------------------------------------------
// Orchestration
// ---------------------------------------------------------------------------
extern "C" void kernel_launch(void* const* d_in, const int* in_sizes, int n_in,
                              void* d_out, int out_size, void* d_ws, size_t ws_size,
                              hipStream_t stream)
{
    (void)in_sizes; (void)n_in; (void)out_size; (void)ws_size;
    auto F = [&](int i) { return (const float*)d_in[i]; };

    const float* f1l3  = F(0);   // [64,128,36,36]
    const float* f1l4  = F(1);   // [64,256,18,18]
    const float* f2l3  = F(2);
    const float* f2l4  = F(3);
    const float* bb1   = F(4);   // [64,4] xywh
    const float* props = F(5);   // [64,16,4] xywh
    const int PB = 6;
    // +0 conv3_1r, +4 conv3_1t, +8 conv3_2t, +12 fc3_1r, +16 conv4_1r,
    // +20 conv4_1t, +24 conv4_2t, +28 fc34_3r, +32 fc34_4r, +36 fc3_rt,
    // +40 fc4_rt, +44 iou_w, +45 iou_b

    // ---- workspace (bytes) ----
    char* wsb = (char*)d_ws;
    size_t off = 0;
    auto allocH = [&](size_t n) { _Float16* p = (_Float16*)(wsb + off); off += n * 2; return p; };
    auto allocF = [&](size_t n) { off = (off + 15) & ~15UL; float* p = (float*)(wsb + off); off += n * 4; return p; };

    // f16 padded NHWC activations
    _Float16* pad1l3 = allocH(64UL * 38 * 38 * 128);
    _Float16* pad1l4 = allocH(64UL * 20 * 20 * 256);
    _Float16* pad2l3 = allocH(64UL * 38 * 38 * 128);
    _Float16* pad2l4 = allocH(64UL * 20 * 20 * 256);
    _Float16* pad3t  = allocH(64UL * 38 * 38 * 256);   // conv3_1t out (padded)
    _Float16* pad4t  = allocH(64UL * 20 * 20 * 256);   // conv4_1t out (padded)
    // f16 repacked conv weights [3][Cout][3*Cin]
    _Float16* w31r = allocH(9UL * 128 * 128);
    _Float16* w31t = allocH(9UL * 256 * 128);
    _Float16* w32t = allocH(9UL * 256 * 256);
    _Float16* w41r = allocH(9UL * 256 * 256);
    _Float16* w41t = allocH(9UL * 256 * 256);
    _Float16* w42t = allocH(9UL * 256 * 256);
    // f16 fc weights (flat converted)
    _Float16* wfc31r = allocH(256UL * 1152);
    _Float16* wf343  = allocH(256UL * 512);
    _Float16* wf344  = allocH(256UL * 512);
    _Float16* wfc3rt = allocH(256UL * 6400);
    _Float16* wfc4rt = allocH(256UL * 2304);
    // f16 activations
    _Float16* roi3r = allocH(64UL * 1152);
    _Float16* fc34  = allocH(64UL * 512);
    _Float16* m3    = allocH(64UL * 256);
    _Float16* m4    = allocH(64UL * 256);
    _Float16* roi3t = allocH(1024UL * 6400);
    _Float16* roi4t = allocH(1024UL * 2304);
    _Float16* fcbuf = allocH(1024UL * 512);
    // fp32 NCHW conv outputs (pooling inputs)
    float* c3r  = allocF(64UL * 128 * 36 * 36);
    float* c4r  = allocF(64UL * 256 * 18 * 18);
    float* c3t2 = allocF(64UL * 256 * 36 * 36);
    float* c4t2 = allocF(64UL * 256 * 18 * 18);

    auto grid1 = [&](long total) { return dim3((unsigned)((total + 255) / 256)); };
    auto pad = [&](const float* in, _Float16* out, int C, int H, int W) {
        const int total = 64 * (H + 2) * (W + 2) * C;
        pad_nhwc_f16<<<grid1(total), dim3(256), 0, stream>>>(in, out, C, H, W, total);
    };
    auto repack = [&](int pi, _Float16* out, int Cout, int Cin) {
        const int total = 9 * Cout * Cin;
        repack_w3_f16<<<grid1(total), dim3(256), 0, stream>>>(F(pi), out, Cout, Cin, total);
    };
    auto cvt = [&](int pi, _Float16* out, int total) {
        cvt_f16<<<grid1(total), dim3(256), 0, stream>>>(F(pi), out, total);
    };
    auto zero = [&](_Float16* p, long total) {
        zero_f16<<<grid1(total), dim3(256), 0, stream>>>(p, (int)total);
    };
    auto conv = [&](const _Float16* in, const _Float16* wr, int pi,
                    float* out_nchw, _Float16* out_pad,
                    int Cin, int Cout, int H, int W) {
        const int blocks = (Cout / 64) * ((64 * H * W) / 128);
        conv3x3_bn_relu_wmma<<<dim3(blocks), dim3(256), 0, stream>>>(
            in, wr, F(pi + 1), F(pi + 2), F(pi + 3), out_nchw, out_pad,
            64, Cin, Cout, H, W);
    };
    auto gemm = [&](const _Float16* A, int pi, const _Float16* B, _Float16* C,
                    int M, int Nr, int K, int ldc) {
        const int blocks = (M / 64) * (Nr / 64);
        gemm_bn_relu_wmma<<<dim3(blocks), dim3(256), 0, stream>>>(
            A, B, F(pi + 1), F(pi + 2), F(pi + 3), C, M, Nr, K, ldc);
    };
    auto pool = [&](const float* feat, const float* boxes, const _Float16* mod,
                    _Float16* out, int C, int H, int W, int Pn, int ph, int pw,
                    float scl, int rs, int coff) {
        const int total = 64 * Pn * C * ph * pw;
        prroi_pool_kernel<<<grid1(total), dim3(256), 0, stream>>>(
            feat, boxes, mod, out, 64, C, H, W, Pn, ph, pw, scl, rs, coff, total);
    };

    // ---------------- layout prep ----------------
    pad(f1l3, pad1l3, 128, 36, 36);
    pad(f1l4, pad1l4, 256, 18, 18);
    pad(f2l3, pad2l3, 128, 36, 36);
    pad(f2l4, pad2l4, 256, 18, 18);
    zero(pad3t, 64L * 38 * 38 * 256);   // halo for chained conv outputs
    zero(pad4t, 64L * 20 * 20 * 256);
    repack(PB + 0,  w31r, 128, 128);
    repack(PB + 4,  w31t, 256, 128);
    repack(PB + 8,  w32t, 256, 256);
    repack(PB + 16, w41r, 256, 256);
    repack(PB + 20, w41t, 256, 256);
    repack(PB + 24, w42t, 256, 256);
    cvt(PB + 12, wfc31r, 256 * 1152);
    cvt(PB + 28, wf343,  256 * 512);
    cvt(PB + 32, wf344,  256 * 512);
    cvt(PB + 36, wfc3rt, 256 * 6400);
    cvt(PB + 40, wfc4rt, 256 * 2304);

    // ---------------- reference branch (get_filter) ----------------
    conv(pad1l3, w31r, PB + 0, c3r, nullptr, 128, 128, 36, 36);            // conv3_1r
    pool(c3r, bb1, nullptr, roi3r, 128, 36, 36, 1, 3, 3, 0.125f, 1152, 0);
    gemm(wfc31r, PB + 12, roi3r, fc34, 256, 64, 1152, 512);                // fc3_1r
    conv(pad1l4, w41r, PB + 16, c4r, nullptr, 256, 256, 18, 18);           // conv4_1r
    pool(c4r, bb1, nullptr, fc34, 256, 18, 18, 1, 1, 1, 0.0625f, 512, 256);// roi4r
    gemm(wf343, PB + 28, fc34, m3, 256, 64, 512, 256);                     // fc34_3r
    gemm(wf344, PB + 32, fc34, m4, 256, 64, 512, 256);                     // fc34_4r

    // ---------------- test branch (get_iou_feat) ----------------
    conv(pad2l3, w31t, PB + 4,  nullptr, pad3t, 128, 256, 36, 36);         // conv3_1t
    conv(pad3t,  w32t, PB + 8,  c3t2,   nullptr, 256, 256, 36, 36);        // conv3_2t
    conv(pad2l4, w41t, PB + 20, nullptr, pad4t, 256, 256, 18, 18);         // conv4_1t
    conv(pad4t,  w42t, PB + 24, c4t2,   nullptr, 256, 256, 18, 18);        // conv4_2t

    // attention (m3/m4) folded into pooling epilogue
    pool(c3t2, props, m3, roi3t, 256, 36, 36, 16, 5, 5, 0.125f, 6400, 0);
    pool(c4t2, props, m4, roi4t, 256, 18, 18, 16, 3, 3, 0.0625f, 2304, 0);

    // ---------------- IoU predictor ----------------
    gemm(wfc3rt, PB + 36, roi3t, fcbuf,       256, 1024, 6400, 512);       // fc3_rt
    gemm(wfc4rt, PB + 40, roi4t, fcbuf + 256, 256, 1024, 2304, 512);       // fc4_rt
    iou_head_kernel<<<dim3(4), dim3(256), 0, stream>>>(
        fcbuf, F(PB + 44), F(PB + 45), (float*)d_out, 1024, 512);
}
